// QGcnn_18348100288873
// MI455X (gfx1250) — matrix operand order repaired and checked by reading, not compiled, not run on hardware
//
#include <hip/hip_runtime.h>

// ---------------------------------------------------------------------------
// CDNA5 (gfx1250) WMMA GNN pipeline.
// wave32; one 32-lane wave per block; one block = one 16-edge tile.
// All matmuls via v_wmma_f32_16x16x32_f16 (f16 A/B, f32 accum).
// ---------------------------------------------------------------------------

typedef __attribute__((ext_vector_type(16))) _Float16 v16h;
typedef __attribute__((ext_vector_type(8)))  _Float16 v8h;
typedef __attribute__((ext_vector_type(8)))  float    v8f;

static __device__ __forceinline__ v8f wmma_f(v16h a, v16h b, v8f c) {
  // (neg_a, A, neg_b, B, c_mod, C, reuse_a, reuse_b)
  return __builtin_amdgcn_wmma_f32_16x16x32_f16(false, a, false, b, (short)0, c, false, false);
}

// A operand (16x32 f16): lane L -> row m = L&15, half = L>>4.
// VGPR 0..3 hold K = 8*half + {0..7} (contiguous), VGPR 4..7 hold K = 16 + 8*half + {0..7}.
// -> two 16B LDS loads per K-chunk kc (each chunk covers 32 K values).
static __device__ __forceinline__ v16h load_A(const _Float16* hrow, int half, int kc) {
  union { v16h v; v8h h[2]; } u;
  u.h[0] = *(const v8h*)(hrow + kc * 32 + 8 * half);
  u.h[1] = *(const v8h*)(hrow + kc * 32 + 16 + 8 * half);
  return u.v;
}

// B operand pre-packed in global scratch: fragment = 256 u32 (= 64 uint4);
// lane L owns u32s [L*8 .. L*8+7] -> two coalesced uint4 loads.
static __device__ __forceinline__ v16h load_B(const uint4* frag, int lane) {
  union { v16h v; uint4 q[2]; } u;
  u.q[0] = frag[lane * 2];
  u.q[1] = frag[lane * 2 + 1];
  return u.v;
}

// ---------------------------------------------------------------------------
// Weight packer: W is [Ldim][K][N] f32 row-major.  Output fragment layout:
// frag f = (l*kcN + kc)*ntN + nt ; within fragment lane L: n = nt*16 + (L&15),
// k0 = kc*32 + (j>>2)*16 + (L>>4)*8 + 2*(j&3) ; u32 j packs f16(W[k0][n]) | f16(W[k0+1][n])<<16.
// Out-of-range (k,n) are zero-padded (handles K=33/48/49, K=16 etc.).
// ---------------------------------------------------------------------------
__global__ void pack_wB(const float* __restrict__ W, int Ldim, int K, int N,
                        int kcN, int ntN, unsigned int* __restrict__ out) {
  int tid = blockIdx.x * blockDim.x + threadIdx.x;
  int total = Ldim * kcN * ntN * 256;
  if (tid >= total) return;
  int j = tid & 7;
  int lane = (tid >> 3) & 31;
  int f = tid >> 8;
  int nt = f % ntN;
  int rest = f / ntN;
  int kc = rest % kcN;
  int l = rest / kcN;
  int n = nt * 16 + (lane & 15);
  int k0 = kc * 32 + ((j >> 2) * 16) + ((lane >> 4) * 8) + 2 * (j & 3);
  float v0 = (k0 < K && n < N) ? W[(size_t)l * K * N + (size_t)k0 * N + n] : 0.f;
  float v1 = (k0 + 1 < K && n < N) ? W[(size_t)l * K * N + (size_t)(k0 + 1) * N + n] : 0.f;
  union { _Float16 h[2]; unsigned int u; } pk;
  pk.h[0] = (_Float16)v0;
  pk.h[1] = (_Float16)v1;
  out[tid] = pk.u;
}

__global__ void zero_f32(float* __restrict__ p, int n) {
  int t = blockIdx.x * blockDim.x + threadIdx.x;
  if (t < n) p[t] = 0.f;
}

// x = agg / max(cnt,1); then rezero agg/cnt for the next NodeConv (idempotent).
__global__ void div_mean(float* __restrict__ agg, float* __restrict__ cnt,
                         float* __restrict__ xout, int n16) {
  int t = blockIdx.x * blockDim.x + threadIdx.x;
  if (t >= n16) return;
  int i = t >> 4;
  float c = cnt[i];                      // all 16 lanes of node i are in one wave
  xout[t] = agg[t] / fmaxf(c, 1.f);
  agg[t] = 0.f;
  if ((t & 15) == 0) cnt[i] = 0.f;       // after lockstep reads above
}

// ---------------------------------------------------------------------------
// NodeConv: per 16-edge tile, X = [x[dst] | x[src] | angle] (16x33, pad 64),
// MLP 33->16 + 5x(16->16), ReLU; atomic scatter-add to agg[dst], cnt[dst].
// ---------------------------------------------------------------------------
__global__ void __launch_bounds__(32) node_conv_kernel(
    const float* __restrict__ x, const int* __restrict__ src, const int* __restrict__ dst,
    const float* __restrict__ angles,
    const uint4* __restrict__ wB0, const float* __restrict__ b0,
    const uint4* __restrict__ wBh, const float* __restrict__ bh,
    float* __restrict__ agg, float* __restrict__ cnt) {
  __shared__ __align__(16) _Float16 hbuf[16][64];
  __shared__ int sdst[16];
  int lane = threadIdx.x;
  int tile = blockIdx.x;
  int n = lane & 15, half = lane >> 4;
  int e = tile * 16 + n;

  int node;
  if (half) { node = src[e]; }
  else      { node = dst[e]; sdst[n] = node; }
  const float* xr = x + (size_t)node * 16;
#pragma unroll
  for (int i = 0; i < 16; i++) hbuf[n][half * 16 + i] = (_Float16)xr[i];
  if (half) {
    hbuf[n][32] = (_Float16)angles[e];
    for (int i = 33; i < 48; i++) hbuf[n][i] = (_Float16)0.f;
  } else {
    v8h z = {};
    *(v8h*)&hbuf[n][48] = z;
    *(v8h*)&hbuf[n][56] = z;
  }
  __syncthreads();

  // input layer: K = 33 (padded to 64) -> 2 k-chunks, 1 n-tile
  v8f c = {};
#pragma unroll
  for (int kc = 0; kc < 2; kc++) {
    v16h a = load_A(hbuf[n], half, kc);
    c = wmma_f(a, load_B(wB0 + kc * 64, lane), c);
  }
  float bias = b0[n];
  __syncthreads();
#pragma unroll
  for (int j = 0; j < 8; j++) hbuf[j + 8 * half][n] = (_Float16)fmaxf(c[j] + bias, 0.f);
  { v8h z = {}; *(v8h*)&hbuf[n][16 + 8 * half] = z; }   // zero K pad 16..31
  __syncthreads();

  v8f cf = {};
  for (int l = 0; l < 5; l++) {
    v8f t = {};
    v16h a = load_A(hbuf[n], half, 0);
    t = wmma_f(a, load_B(wBh + l * 64, lane), t);
    float bl = bh[l * 16 + n];
    __syncthreads();
    if (l < 4) {
#pragma unroll
      for (int j = 0; j < 8; j++) hbuf[j + 8 * half][n] = (_Float16)fmaxf(t[j] + bl, 0.f);
    } else {
#pragma unroll
      for (int j = 0; j < 8; j++) cf[j] = fmaxf(t[j] + bl, 0.f);
    }
    __syncthreads();
  }

  // scatter: lane holds rows r=j+8*half, col n
#pragma unroll
  for (int j = 0; j < 8; j++) {
    int r = j + 8 * half;
    atomicAdd(&agg[(size_t)sdst[r] * 16 + n], cf[j]);
  }
  if (half == 0) atomicAdd(&cnt[sdst[n]], 1.0f);
}

// ---------------------------------------------------------------------------
// EdgeConv pair-MLP: 2C=32 -> H=48, 5x(48->48 padded to K=64), ReLU.
// Outputs 16x48 tile as 3 v8f accumulators.
// ---------------------------------------------------------------------------
static __device__ __forceinline__ void run_edge_mlp(
    _Float16 (*hbuf)[64], const float* __restrict__ x,
    const int* nodesFirst, const int* nodesSecond,
    const uint4* __restrict__ wB0, const float* __restrict__ b0,
    const uint4* __restrict__ wBh, const float* __restrict__ bh,
    int lane, v8f& o0, v8f& o1, v8f& o2) {
  int n = lane & 15, half = lane >> 4;

  // stage X = [x[first] | x[second]] into cols 0..31
  const float* xr = x + (size_t)(half ? nodesSecond[n] : nodesFirst[n]) * 16;
#pragma unroll
  for (int i = 0; i < 16; i++) hbuf[n][half * 16 + i] = (_Float16)xr[i];
  __syncthreads();

  // layer0: K=32 exact -> 1 k-chunk, 3 n-tiles
  v8f c0 = {}, c1 = {}, c2 = {};
  {
    v16h a = load_A(hbuf[n], half, 0);
    c0 = wmma_f(a, load_B(wB0 + 0 * 64, lane), c0);
    c1 = wmma_f(a, load_B(wB0 + 1 * 64, lane), c1);
    c2 = wmma_f(a, load_B(wB0 + 2 * 64, lane), c2);
  }
  float q0 = b0[n], q1 = b0[16 + n], q2 = b0[32 + n];
  __syncthreads();
#pragma unroll
  for (int j = 0; j < 8; j++) {
    int r = j + 8 * half;
    hbuf[r][n]      = (_Float16)fmaxf(c0[j] + q0, 0.f);
    hbuf[r][16 + n] = (_Float16)fmaxf(c1[j] + q1, 0.f);
    hbuf[r][32 + n] = (_Float16)fmaxf(c2[j] + q2, 0.f);
  }
  __syncthreads();

  // 5 hidden layers: K=48 padded to 64 -> 2 k-chunks, 3 n-tiles
  for (int l = 0; l < 5; l++) {
    v8f d0 = {}, d1 = {}, d2 = {};
#pragma unroll
    for (int kc = 0; kc < 2; kc++) {
      v16h a = load_A(hbuf[n], half, kc);
      const uint4* base = wBh + (size_t)((l * 2 + kc) * 3) * 64;
      d0 = wmma_f(a, load_B(base + 0 * 64, lane), d0);
      d1 = wmma_f(a, load_B(base + 1 * 64, lane), d1);
      d2 = wmma_f(a, load_B(base + 2 * 64, lane), d2);
    }
    float p0 = bh[l * 48 + n], p1 = bh[l * 48 + 16 + n], p2 = bh[l * 48 + 32 + n];
    __syncthreads();
    if (l < 4) {
#pragma unroll
      for (int j = 0; j < 8; j++) {
        int r = j + 8 * half;
        hbuf[r][n]      = (_Float16)fmaxf(d0[j] + p0, 0.f);
        hbuf[r][16 + n] = (_Float16)fmaxf(d1[j] + p1, 0.f);
        hbuf[r][32 + n] = (_Float16)fmaxf(d2[j] + p2, 0.f);
      }
    } else {
#pragma unroll
      for (int j = 0; j < 8; j++) {
        d0[j] = fmaxf(d0[j] + p0, 0.f);
        d1[j] = fmaxf(d1[j] + p1, 0.f);
        d2[j] = fmaxf(d2[j] + p2, 0.f);
      }
      o0 = d0; o1 = d1; o2 = d2;
    }
    __syncthreads();
  }
}

// mode 1: combine concat([fe, actions]) @ wc(49x48) via WMMA -> ef_out (f16, 48ch)
// mode 2: combine concat([fe, ef_in]) @ wc2(96x1) via VALU dot -> out1 (f32, 1ch)
__global__ void __launch_bounds__(32) edge_conv_kernel(
    const float* __restrict__ x, const int* __restrict__ src, const int* __restrict__ dst,
    const uint4* __restrict__ wB0, const float* __restrict__ b0,
    const uint4* __restrict__ wBh, const float* __restrict__ bh,
    int mode,
    const uint4* __restrict__ wBc, const float* __restrict__ bcv,
    const float* __restrict__ actions,
    const _Float16* __restrict__ ef_in,
    const float* __restrict__ wc2, const float* __restrict__ bc2,
    _Float16* __restrict__ ef_out, float* __restrict__ out1,
    float* __restrict__ sl_partial) {
  __shared__ __align__(16) _Float16 hbuf[16][64];
  __shared__ float fbuf[16][96];
  __shared__ int sA[16], sB[16];
  int lane = threadIdx.x;
  int tile = blockIdx.x;
  int n = lane & 15, half = lane >> 4;
  int e_n = tile * 16 + n;

  if (half) sB[n] = src[e_n];
  else      sA[n] = dst[e_n];
  { v8h z = {}; *(v8h*)&hbuf[n][48 + 8 * half] = z; }   // K pad 48..63 stays zero
  __syncthreads();

  v8f f10, f11, f12, f20, f21, f22;
  run_edge_mlp(hbuf, x, sA, sB, wB0, b0, wBh, bh, lane, f10, f11, f12); // [x_dst,x_src]
  run_edge_mlp(hbuf, x, sB, sA, wB0, b0, wBh, bh, lane, f20, f21, f22); // [x_src,x_dst]

  float sld = 0.f;
  v8f e0, e1, e2;
#pragma unroll
  for (int j = 0; j < 8; j++) {
    float d0 = f10[j] - f20[j], d1 = f11[j] - f21[j], d2 = f12[j] - f22[j];
    sld += d0 * d0 + d1 * d1 + d2 * d2;
    e0[j] = 0.5f * (f10[j] + f20[j]);
    e1[j] = 0.5f * (f11[j] + f21[j]);
    e2[j] = 0.5f * (f12[j] + f22[j]);
  }
#pragma unroll
  for (int off = 16; off > 0; off >>= 1) sld += __shfl_xor(sld, off, 32);
  if (lane == 0) sl_partial[tile] = sld;

  if (mode == 1) {
    __syncthreads();
#pragma unroll
    for (int j = 0; j < 8; j++) {
      int r = j + 8 * half;
      hbuf[r][n]      = (_Float16)e0[j];
      hbuf[r][16 + n] = (_Float16)e1[j];
      hbuf[r][32 + n] = (_Float16)e2[j];
    }
    if (half == 0) hbuf[n][48] = (_Float16)actions[e_n];   // col 48; 49..63 still zero
    __syncthreads();
    v8f c0 = {}, c1 = {}, c2 = {};
#pragma unroll
    for (int kc = 0; kc < 2; kc++) {
      v16h a = load_A(hbuf[n], half, kc);
      const uint4* base = wBc + (size_t)(kc * 3) * 64;
      c0 = wmma_f(a, load_B(base + 0 * 64, lane), c0);
      c1 = wmma_f(a, load_B(base + 1 * 64, lane), c1);
      c2 = wmma_f(a, load_B(base + 2 * 64, lane), c2);
    }
    float q0 = bcv[n], q1 = bcv[16 + n], q2 = bcv[32 + n];
    __syncthreads();
#pragma unroll
    for (int j = 0; j < 8; j++) {     // linear output, no ReLU
      int r = j + 8 * half;
      hbuf[r][n]      = (_Float16)(c0[j] + q0);
      hbuf[r][16 + n] = (_Float16)(c1[j] + q1);
      hbuf[r][32 + n] = (_Float16)(c2[j] + q2);
    }
    __syncthreads();
    // coalesced f16 row copy: lane covers 48B of row n
    const uint4* sp = (const uint4*)&hbuf[n][half * 24];
    uint4* dp = (uint4*)((char*)ef_out + (size_t)e_n * 96 + (size_t)half * 48);
    dp[0] = sp[0]; dp[1] = sp[1]; dp[2] = sp[2];
  } else {
    __syncthreads();
#pragma unroll
    for (int j = 0; j < 8; j++) {
      int r = j + 8 * half;
      fbuf[r][n] = e0[j]; fbuf[r][16 + n] = e1[j]; fbuf[r][32 + n] = e2[j];
    }
    const _Float16* er = ef_in + (size_t)e_n * 48 + half * 24;
#pragma unroll
    for (int i = 0; i < 24; i++) fbuf[n][48 + half * 24 + i] = (float)er[i];
    __syncthreads();
    if (half == 0) {
      float s = bc2[0];
      for (int i = 0; i < 96; i++) s += fbuf[n][i] * wc2[i];
      out1[e_n] = s;
    }
  }
}

__global__ void reduce_sl(const float* __restrict__ p1, const float* __restrict__ p2,
                          int ntiles, int nE, float* __restrict__ out) {
  __shared__ float s[256];
  float a = 0.f;
  for (int i = threadIdx.x; i < ntiles; i += 256) a += p1[i] + p2[i];
  s[threadIdx.x] = a;
  __syncthreads();
  for (int st = 128; st > 0; st >>= 1) {
    if ((int)threadIdx.x < st) s[threadIdx.x] += s[threadIdx.x + st];
    __syncthreads();
  }
  if (threadIdx.x == 0) out[0] = s[0] / (2.0f * (float)nE * 48.0f);
}

// ---------------------------------------------------------------------------
extern "C" void kernel_launch(void* const* d_in, const int* in_sizes, int n_in,
                              void* d_out, int out_size, void* d_ws, size_t ws_size,
                              hipStream_t stream) {
  (void)n_in; (void)out_size; (void)ws_size;

  const float* node_features = (const float*)d_in[0];
  const int*   edge_index    = (const int*)d_in[1];
  const float* angles        = (const float*)d_in[2];
  const float* actions       = (const float*)d_in[4];
  const float* nc1_w0 = (const float*)d_in[5],  *nc1_b0 = (const float*)d_in[6];
  const float* nc1_wh = (const float*)d_in[7],  *nc1_bh = (const float*)d_in[8];
  const float* nc2_w0 = (const float*)d_in[9],  *nc2_b0 = (const float*)d_in[10];
  const float* nc2_wh = (const float*)d_in[11], *nc2_bh = (const float*)d_in[12];
  const float* ec1_w0 = (const float*)d_in[13], *ec1_b0 = (const float*)d_in[14];
  const float* ec1_wh = (const float*)d_in[15], *ec1_bh = (const float*)d_in[16];
  const float* ec1_wc = (const float*)d_in[17], *ec1_bc = (const float*)d_in[18];
  const float* ec2_w0 = (const float*)d_in[19], *ec2_b0 = (const float*)d_in[20];
  const float* ec2_wh = (const float*)d_in[21], *ec2_bh = (const float*)d_in[22];
  const float* ec2_wc = (const float*)d_in[23], *ec2_bc = (const float*)d_in[24];

  const int nE = in_sizes[2];          // 600000
  const int nN = in_sizes[0] / 16;     // 100000
  const int tiles = nE / 16;           // 37500 (nE % 16 == 0)
  const int* srcp = edge_index;
  const int* dstp = edge_index + nE;

  char* ws = (char*)d_ws;
  size_t off = 0;
  auto alloc = [&](size_t bytes) -> char* {
    char* p = ws + off;
    off += (bytes + 255) & ~(size_t)255;
    return p;
  };
  uint4* wB_nc1_w0 = (uint4*)alloc(512 * 4);     // K=33->kc2, nt1
  uint4* wB_nc1_wh = (uint4*)alloc(1280 * 4);    // 5x K=16->kc1, nt1
  uint4* wB_nc2_w0 = (uint4*)alloc(512 * 4);
  uint4* wB_nc2_wh = (uint4*)alloc(1280 * 4);
  uint4* wB_ec1_w0 = (uint4*)alloc(768 * 4);     // K=32->kc1, nt3
  uint4* wB_ec1_wh = (uint4*)alloc(7680 * 4);    // 5x K=48->kc2, nt3
  uint4* wB_ec1_wc = (uint4*)alloc(1536 * 4);    // K=49->kc2, nt3
  uint4* wB_ec2_w0 = (uint4*)alloc(768 * 4);
  uint4* wB_ec2_wh = (uint4*)alloc(7680 * 4);
  float* agg = (float*)alloc((size_t)nN * 16 * 4);
  float* cnt = (float*)alloc((size_t)nN * 4);
  float* x1  = (float*)alloc((size_t)nN * 16 * 4);
  float* x2  = (float*)alloc((size_t)nN * 16 * 4);
  _Float16* ef = (_Float16*)alloc((size_t)nE * 48 * 2);
  float* p1  = (float*)alloc((size_t)tiles * 4);
  float* p2  = (float*)alloc((size_t)tiles * 4);

  float* out = (float*)d_out;

  // --- pack all weights into per-lane B-fragment layout ---
  auto pack = [&](const float* W, int L, int K, int N, int kcN, int ntN, void* dstb) {
    int total = L * kcN * ntN * 256;
    pack_wB<<<(total + 255) / 256, 256, 0, stream>>>(W, L, K, N, kcN, ntN, (unsigned int*)dstb);
  };
  pack(nc1_w0, 1, 33, 16, 2, 1, wB_nc1_w0);
  pack(nc1_wh, 5, 16, 16, 1, 1, wB_nc1_wh);
  pack(nc2_w0, 1, 33, 16, 2, 1, wB_nc2_w0);
  pack(nc2_wh, 5, 16, 16, 1, 1, wB_nc2_wh);
  pack(ec1_w0, 1, 32, 48, 1, 3, wB_ec1_w0);
  pack(ec1_wh, 5, 48, 48, 2, 3, wB_ec1_wh);
  pack(ec1_wc, 1, 49, 48, 2, 3, wB_ec1_wc);
  pack(ec2_w0, 1, 32, 48, 1, 3, wB_ec2_w0);
  pack(ec2_wh, 5, 48, 48, 2, 3, wB_ec2_wh);

  zero_f32<<<(nN * 16 + 255) / 256, 256, 0, stream>>>(agg, nN * 16);
  zero_f32<<<(nN + 255) / 256, 256, 0, stream>>>(cnt, nN);

  // NodeConv1 -> x1
  node_conv_kernel<<<tiles, 32, 0, stream>>>(node_features, srcp, dstp, angles,
                                             wB_nc1_w0, nc1_b0, wB_nc1_wh, nc1_bh, agg, cnt);
  div_mean<<<(nN * 16 + 255) / 256, 256, 0, stream>>>(agg, cnt, x1, nN * 16);

  // EdgeConv1 -> ef (f16), sl1 partials
  edge_conv_kernel<<<tiles, 32, 0, stream>>>(x1, srcp, dstp,
                                             wB_ec1_w0, ec1_b0, wB_ec1_wh, ec1_bh,
                                             1, wB_ec1_wc, ec1_bc, actions,
                                             (const _Float16*)nullptr, nullptr, nullptr,
                                             ef, nullptr, p1);

  // NodeConv2 -> x2
  node_conv_kernel<<<tiles, 32, 0, stream>>>(x1, srcp, dstp, angles,
                                             wB_nc2_w0, nc2_b0, wB_nc2_wh, nc2_bh, agg, cnt);
  div_mean<<<(nN * 16 + 255) / 256, 256, 0, stream>>>(agg, cnt, x2, nN * 16);

  // EdgeConv2 -> d_out[0..nE), sl2 partials
  edge_conv_kernel<<<tiles, 32, 0, stream>>>(x2, srcp, dstp,
                                             wB_ec2_w0, ec2_b0, wB_ec2_wh, ec2_bh,
                                             2, (const uint4*)nullptr, nullptr, nullptr,
                                             ef, ec2_wc, ec2_bc,
                                             (_Float16*)nullptr, out, p2);

  // side_loss -> d_out[nE]
  reduce_sl<<<1, 256, 0, stream>>>(p1, p2, tiles, nE, out + nE);
}
